// AttModel_16149077033206
// MI455X (gfx1250) — matrix operand order repaired and checked by reading, not compiled, not run on hardware
//
#include <hip/hip_runtime.h>
#include <hip/hip_bf16.h>
#include <math.h>

typedef _Float16 half_t;
typedef __attribute__((ext_vector_type(16))) _Float16 v16h;
typedef __attribute__((ext_vector_type(8)))  _Float16 v8h;
typedef __attribute__((ext_vector_type(8)))  float    v8f;

namespace {

constexpr int kInFeat  = 66;
constexpr int kDModel  = 512;
constexpr int kDctN    = 20;
constexpr int kHid     = 512;
constexpr int kGcnIn   = 40;   // 2*DCT_N
constexpr int kBS      = 512;
constexpr int kInputN  = 50;
constexpr int kVL      = 35;   // KS + output_n
constexpr int kVN      = 16;   // input_n - KS - output_n + 1
constexpr long kRowsGcn = (long)kBS * kInFeat;  // 33792

// ---------------- DCT table (orthonormal DCT-II; inverse == transpose) -------
__global__ void init_dct_kernel(float* dctT) {
  for (int t = threadIdx.x; t < kDctN * kVL; t += blockDim.x) {
    int k = t / kVL, v = t % kVL;
    float w = (k == 0) ? sqrtf(1.0f / kVL) : sqrtf(2.0f / kVL);
    dctT[t] = w * cosf(3.14159265358979323846f * (v + 0.5f) * k / kVL);
  }
}

// ---------------- f32 -> f16 convert ----------------------------------------
__global__ void cvt_f16_kernel(const float* __restrict__ in, half_t* __restrict__ out,
                               long n, float scale) {
  long i = (long)blockIdx.x * blockDim.x + threadIdx.x;
  if (i < n) out[i] = (half_t)(in[i] * scale);
}

// conv1 weight (O=512, I=66, T=6) -> B[(t*66+i)][o]   (folds 1/1000 scale)
__global__ void reorder_w6_kernel(const float* __restrict__ W, half_t* __restrict__ B,
                                  float scale) {
  long i = (long)blockIdx.x * blockDim.x + threadIdx.x;
  if (i >= (long)kDModel * kInFeat * 6) return;
  int o = (int)(i / (kInFeat * 6));
  int r = (int)(i % (kInFeat * 6));
  int ii = r / 6, t = r % 6;
  B[(long)(t * kInFeat + ii) * kDModel + o] = (half_t)(W[i] * scale);
}

// conv2 weight (O=512, M=512, T=5) -> B[(t*512+m)][o]
__global__ void reorder_w5_kernel(const float* __restrict__ W, half_t* __restrict__ B) {
  long i = (long)blockIdx.x * blockDim.x + threadIdx.x;
  if (i >= (long)kDModel * kDModel * 5) return;
  int o = (int)(i / (kDModel * 5));
  int r = (int)(i % (kDModel * 5));
  int m = r / 5, t = r % 5;
  B[(long)(t * kDModel + m) * kDModel + o] = (half_t)W[i];
}

// ---------------- Generic WMMA GEMM ------------------------------------------
// C[row,col] = act( sum_k A[row,k] * B[k,col] ), row start(row) =
// baseOff + (row/P)*batchStride + (row%P)*rowStride  (strided-view im2col).
// Block tile 128x64, wave tile 32x32 (2x2 wmma accums), K-step 64,
// double-buffered LDS (one barrier per stage), register prefetch.
// Rows multiple of 128 at every call site; N guarded on store.
template <bool RELU>
__global__ __launch_bounds__(256) void gemm_f16_kernel(
    const half_t* __restrict__ A, long baseOff, int P, long rowStride, long batchStride,
    int K, const half_t* __restrict__ B, int ldb,
    half_t* __restrict__ C, int ldc, int N) {
  __shared__ __align__(16) half_t sA[2][128][72];  // 128 rows x 64 k (+8 pad)
  __shared__ __align__(16) half_t sB[2][64][72];   // 64 k x 64 n (+8 pad)

  const int tid  = threadIdx.x;
  const int lane = tid & 31;
  const int wave = tid >> 5;
  const int wm   = wave >> 1;  // 0..3 -> 32-row band
  const int wn   = wave & 1;   // 0..1 -> 32-col band
  const long rowBase = (long)blockIdx.x * 128;
  const int  colBase = blockIdx.y * 64;

  v8f acc[2][2] = {};

  // A loader: row ar = tid/2 (0..127), cols ac = (tid&1)*32 .. +32
  const int ar = tid >> 1;
  const int ac = (tid & 1) << 5;
  const long aRow = rowBase + ar;
  const half_t* aBase = A + baseOff + (aRow / P) * batchStride + (aRow % P) * rowStride;
  const uint32_t* a32 = (const uint32_t*)aBase;  // 4B aligned at all call sites
  // B loader: row br = tid/4 (0..63), cols bc = (tid&3)*16 .. +16
  const int br = tid >> 2;
  const int bc = (tid & 3) << 4;
  const half_t* bBase = B + colBase + bc;        // 16B aligned rows

  uint32_t aw[16];
  v8h bl0, bl1;

  auto loadA = [&](int k0, bool guard) {
    if (!guard) {
      const int base = (k0 + ac) >> 1;
      #pragma unroll
      for (int j = 0; j < 16; ++j) aw[j] = a32[base + j];
    } else {
      #pragma unroll
      for (int j = 0; j < 16; ++j) {
        const int kg = k0 + ac + 2 * j;
        const int kc0 = (kg < K) ? kg : (K - 1);      // clamped: always in-bounds
        const int kc1 = (kg + 1 < K) ? (kg + 1) : (K - 1);
        uint32_t lo = *(const uint16_t*)(aBase + kc0);
        uint32_t hi = *(const uint16_t*)(aBase + kc1);
        if (kg >= K) lo = 0u;
        if (kg + 1 >= K) hi = 0u;
        aw[j] = lo | (hi << 16);
      }
    }
  };

  auto loadB = [&](int k0, bool guard) {
    const int kg = k0 + br;
    const long kr = (!guard || kg < K) ? (long)kg : (long)(K - 1);
    bl0 = *(const v8h*)(bBase + kr * ldb);
    bl1 = *(const v8h*)(bBase + kr * ldb + 8);
    if (guard && kg >= K) {
      #pragma unroll
      for (int j = 0; j < 8; ++j) { bl0[j] = (half_t)0.0f; bl1[j] = (half_t)0.0f; }
    }
  };

  auto storeLDS = [&](int buf) {
    #pragma unroll
    for (int q = 0; q < 4; ++q) {
      uint4 u = {aw[4 * q], aw[4 * q + 1], aw[4 * q + 2], aw[4 * q + 3]};
      *(uint4*)&sA[buf][ar][ac + 8 * q] = u;
    }
    *(v8h*)&sB[buf][br][bc] = bl0;
    *(v8h*)&sB[buf][br][bc + 8] = bl1;
  };

  auto compute = [&](int buf) {
    const int m = lane & 15;
    const int g = lane >> 4;
    #pragma unroll
    for (int ks = 0; ks < 64; ks += 32) {
      v16h af[2], bf[2];
      #pragma unroll
      for (int mi = 0; mi < 2; ++mi) {
        v8h lo = *(const v8h*)&sA[buf][32 * wm + 16 * mi + m][ks + g * 8];
        v8h hi = *(const v8h*)&sA[buf][32 * wm + 16 * mi + m][ks + 16 + g * 8];
        #pragma unroll
        for (int j = 0; j < 8; ++j) { af[mi][j] = lo[j]; af[mi][8 + j] = hi[j]; }
      }
      #pragma unroll
      for (int ni = 0; ni < 2; ++ni) {
        #pragma unroll
        for (int j = 0; j < 8; ++j) {
          bf[ni][j] = sB[buf][ks + g * 8 + j][32 * wn + 16 * ni + m];
          bf[ni][8 + j] = sB[buf][ks + 16 + g * 8 + j][32 * wn + 16 * ni + m];
        }
      }
      #pragma unroll
      for (int mi = 0; mi < 2; ++mi) {
        #pragma unroll
        for (int ni = 0; ni < 2; ++ni) {
          acc[mi][ni] = __builtin_amdgcn_wmma_f32_16x16x32_f16(
              false, af[mi], false, bf[ni], (short)0, acc[mi][ni], false, false);
        }
      }
    }
  };

  const int kFull = K & ~63;
  const bool hasTail = (kFull < K);
  const int nStages = (kFull >> 6) + (hasTail ? 1 : 0);
  const int lastS = nStages - 1;

  loadA(0, hasTail && lastS == 0);
  loadB(0, hasTail && lastS == 0);
  storeLDS(0);
  for (int s = 0; s < nStages; ++s) {
    const int buf = s & 1;
    if (s + 1 < nStages) {  // prefetch next stage into registers
      const bool g = hasTail && (s + 1 == lastS);
      loadA((s + 1) << 6, g);
      loadB((s + 1) << 6, g);
    }
    __syncthreads();        // buf ready (stores from previous iteration complete)
    compute(buf);
    if (s + 1 < nStages) storeLDS(buf ^ 1);
  }

  // C layout: VGPR rr -> M = rr + (lane<16 ? 0 : 8); N = lane%16
  const int n  = lane & 15;
  const int mb = (lane >> 4) << 3;
  #pragma unroll
  for (int mi = 0; mi < 2; ++mi) {
    const long rowOut = rowBase + 32 * wm + 16 * mi + mb;
    #pragma unroll
    for (int ni = 0; ni < 2; ++ni) {
      const int cc = colBase + 32 * wn + 16 * ni + n;
      if (cc < N) {
        #pragma unroll
        for (int rr = 0; rr < 8; ++rr) {
          float v = acc[mi][ni][rr];
          if (RELU) v = fmaxf(v, 0.0f);
          C[(rowOut + rr) * ldc + cc] = (half_t)v;
        }
      }
    }
  }
}

// ---------------- attention scores + normalization ---------------------------
__global__ __launch_bounds__(512) void score_att_kernel(const half_t* __restrict__ q,
                                                        const half_t* __restrict__ key,
                                                        float* __restrict__ att) {
  const int b = blockIdx.x;
  const int lane = threadIdx.x & 31;
  const int w = threadIdx.x >> 5;  // p = w
  const half_t* qb = q + (long)b * kDModel;
  const half_t* kb = key + ((long)b * kVN + w) * kDModel;
  float s = 0.0f;
  for (int o = lane; o < kDModel; o += 32) s += (float)qb[o] * (float)kb[o];
  #pragma unroll
  for (int off = 16; off > 0; off >>= 1) s += __shfl_down(s, off, 32);
  __shared__ float sc[kVN];
  __shared__ float tot;
  if (lane == 0) sc[w] = s + 1e-15f;
  __syncthreads();
  if (threadIdx.x == 0) {
    float t = 0.0f;
    for (int i = 0; i < kVN; ++i) t += sc[i];
    tot = t;
  }
  __syncthreads();
  if (threadIdx.x < kVN) att[b * kVN + threadIdx.x] = sc[threadIdx.x] / tot;
}

// ---------------- fused dct_att + dct_in -> x (f32 and f16) ------------------
__global__ __launch_bounds__(256) void build_x_kernel(const float* __restrict__ src,
                                                      const float* __restrict__ att,
                                                      const float* __restrict__ dctT,
                                                      float* __restrict__ xf,
                                                      half_t* __restrict__ xh) {
  const int b = blockIdx.x;
  __shared__ float sAtt[kVN];
  __shared__ float s2[kVL][kInFeat];
  __shared__ float sTail[10][kInFeat];
  if (threadIdx.x < kVN) sAtt[threadIdx.x] = att[b * kVN + threadIdx.x];
  for (int i = threadIdx.x; i < 10 * kInFeat; i += 256) {
    int v = i / kInFeat, f = i % kInFeat;
    sTail[v][f] = src[((long)b * kInputN + 40 + v) * kInFeat + f];
  }
  __syncthreads();
  for (int i = threadIdx.x; i < kVL * kInFeat; i += 256) {
    int v = i / kInFeat, f = i % kInFeat;
    float a = 0.0f;
    #pragma unroll
    for (int n = 0; n < kVN; ++n) a += sAtt[n] * src[((long)b * kInputN + n + v) * kInFeat + f];
    s2[v][f] = a;
  }
  __syncthreads();
  for (int i = threadIdx.x; i < kInFeat * kDctN; i += 256) {
    int f = i / kDctN, k = i % kDctN;
    float da = 0.0f, di = 0.0f;
    #pragma unroll
    for (int v = 0; v < kVL; ++v) {
      float d = dctT[k * kVL + v];
      da += d * s2[v][f];
      di += d * sTail[(v < 10) ? v : 9][f];  // idx2 = [40..49, 49 x 25]
    }
    long o = ((long)b * kInFeat + f) * kGcnIn;
    xf[o + k] = di;          xf[o + kDctN + k] = da;
    xh[o + k] = (half_t)di;  xh[o + kDctN + k] = (half_t)da;
  }
}

// ---------------- GCN adjacency + bias + bn + tanh (+residual), LDS tiled ----
__global__ __launch_bounds__(256) void gcn_att_kernel(
    const half_t* __restrict__ C, const float* __restrict__ Aatt,
    const float* __restrict__ bias, const float* __restrict__ gam,
    const float* __restrict__ bet, const half_t* __restrict__ resid,
    half_t* __restrict__ y, float bnscale) {
  const int b = blockIdx.y;
  const int h0 = blockIdx.x * 128;
  __shared__ float sC[kInFeat][128];
  __shared__ float sA2[kInFeat * kInFeat];
  const half_t* Cb = C + (long)b * kInFeat * kHid + h0;
  for (int i = threadIdx.x; i < kInFeat * 128; i += 256)
    sC[i >> 7][i & 127] = (float)Cb[(long)(i >> 7) * kHid + (i & 127)];
  for (int i = threadIdx.x; i < kInFeat * kInFeat; i += 256) sA2[i] = Aatt[i];
  __syncthreads();
  for (int i = threadIdx.x; i < kInFeat * 128; i += 256) {
    const int n = i >> 7, hl = i & 127, h = h0 + hl;
    float acc = bias[h];
    const float* Ar = sA2 + n * kInFeat;
    #pragma unroll 6
    for (int m = 0; m < kInFeat; ++m) acc += Ar[m] * sC[m][hl];
    const int idx = n * kHid + h;
    acc = tanhf(acc * (gam[idx] * bnscale) + bet[idx]);
    const long o = ((long)b * kInFeat + n) * kHid + h;
    if (resid) acc += (float)resid[o];
    y[o] = (half_t)acc;
  }
}

// ---------------- final GCN (gc7): A@(yW) + b + x -> dct_out f32, LDS tiled --
__global__ __launch_bounds__(256) void gcn_final_kernel(const half_t* __restrict__ C7,
                                                        const float* __restrict__ Aatt,
                                                        const float* __restrict__ bias,
                                                        const float* __restrict__ xf32,
                                                        float* __restrict__ dct_out) {
  const int b = blockIdx.x;
  __shared__ float sC[kInFeat][kGcnIn];
  __shared__ float sA2[kInFeat * kInFeat];
  const half_t* Cb = C7 + (long)b * kInFeat * kGcnIn;
  for (int i = threadIdx.x; i < kInFeat * kGcnIn; i += 256)
    sC[i / kGcnIn][i % kGcnIn] = (float)Cb[i];
  for (int i = threadIdx.x; i < kInFeat * kInFeat; i += 256) sA2[i] = Aatt[i];
  __syncthreads();
  for (int i = threadIdx.x; i < kInFeat * kGcnIn; i += 256) {
    const int n = i / kGcnIn, k = i % kGcnIn;
    float acc = bias[k];
    const float* Ar = sA2 + n * kInFeat;
    #pragma unroll 6
    for (int m = 0; m < kInFeat; ++m) acc += Ar[m] * sC[m][k];
    const long idx = ((long)b * kInFeat + n) * kGcnIn + k;
    dct_out[idx] = acc + xf32[idx];
  }
}

// ---------------- inverse DCT (idct = dct^T) ---------------------------------
__global__ __launch_bounds__(256) void idct_kernel(const float* __restrict__ dct_out,
                                                   const float* __restrict__ dctT,
                                                   float* __restrict__ out) {
  const int b = blockIdx.x;
  __shared__ float sD[kInFeat][kDctN];
  for (int i = threadIdx.x; i < kInFeat * kDctN; i += 256) {
    sD[i / kDctN][i % kDctN] = dct_out[((long)b * kInFeat + i / kDctN) * kGcnIn + i % kDctN];
  }
  __syncthreads();
  for (int i = threadIdx.x; i < kVL * kInFeat; i += 256) {
    int v = i / kInFeat, f = i % kInFeat;
    float a = 0.0f;
    #pragma unroll
    for (int k = 0; k < kDctN; ++k) a += dctT[k * kVL + v] * sD[f][k];
    out[(long)b * kVL * kInFeat + (long)v * kInFeat + f] = a;
  }
}

}  // namespace

extern "C" void kernel_launch(void* const* d_in, const int* in_sizes, int n_in,
                              void* d_out, int out_size, void* d_ws, size_t ws_size,
                              hipStream_t stream) {
  (void)in_sizes; (void)n_in; (void)out_size; (void)ws_size;
  const float* src    = (const float*)d_in[0];
  const float* Wq1    = (const float*)d_in[1];
  const float* Wq2    = (const float*)d_in[2];
  const float* Wk1    = (const float*)d_in[3];
  const float* Wk2    = (const float*)d_in[4];
  const float* gc1_W  = (const float*)d_in[5];
  const float* gc1_at = (const float*)d_in[6];
  const float* gc1_b  = (const float*)d_in[7];
  const float* bn1_g  = (const float*)d_in[8];
  const float* bn1_b  = (const float*)d_in[9];
  const float* gcb_W  = (const float*)d_in[10];
  const float* gcb_at = (const float*)d_in[11];
  const float* gcb_b  = (const float*)d_in[12];
  const float* gcb_g  = (const float*)d_in[13];
  const float* gcb_be = (const float*)d_in[14];
  const float* gc7_W  = (const float*)d_in[15];
  const float* gc7_at = (const float*)d_in[16];
  const float* gc7_b  = (const float*)d_in[17];
  float* out = (float*)d_out;

  char* wp = (char*)d_ws;
  auto alloc = [&](size_t bytes) {
    char* ret = wp;
    wp += (bytes + 255) & ~(size_t)255;
    return ret;
  };
  float*  dctT  = (float*)alloc(kDctN * kVL * sizeof(float));
  half_t* src16 = (half_t*)alloc((size_t)kBS * kInputN * kInFeat * 2);
  half_t* Bk1   = (half_t*)alloc((size_t)396 * 512 * 2);
  half_t* Bq1   = (half_t*)alloc((size_t)396 * 512 * 2);
  half_t* Bk2   = (half_t*)alloc((size_t)2560 * 512 * 2);
  half_t* Bq2   = (half_t*)alloc((size_t)2560 * 512 * 2);
  half_t* Wg1   = (half_t*)alloc((size_t)kGcnIn * kHid * 2);
  half_t* Wgb   = (half_t*)alloc((size_t)4 * kHid * kHid * 2);
  half_t* Wg7   = (half_t*)alloc((size_t)kHid * kGcnIn * 2);
  half_t* out1k = (half_t*)alloc((size_t)kBS * 20 * 512 * 2);
  half_t* out1q = (half_t*)alloc((size_t)kBS * 5 * 512 * 2);
  half_t* keyf  = (half_t*)alloc((size_t)kBS * 16 * 512 * 2);
  half_t* qf    = (half_t*)alloc((size_t)kBS * 512 * 2);
  float*  attb  = (float*)alloc((size_t)kBS * kVN * sizeof(float));
  float*  xf32  = (float*)alloc((size_t)kRowsGcn * kGcnIn * sizeof(float));
  half_t* xf16  = (half_t*)alloc((size_t)kRowsGcn * kGcnIn * 2);
  half_t* ybuf  = (half_t*)alloc((size_t)kRowsGcn * kHid * 2);
  half_t* y0buf = (half_t*)alloc((size_t)kRowsGcn * kHid * 2);
  half_t* Cbuf  = (half_t*)alloc((size_t)kRowsGcn * kHid * 2);
  half_t* C7    = (half_t*)alloc((size_t)kRowsGcn * kGcnIn * 2);
  float*  dcto  = (float*)alloc((size_t)kRowsGcn * kGcnIn * sizeof(float));

  const float bnscale = 1.0f / sqrtf(1.0f + 1e-5f);

  // ---- prep: DCT table + weight conversions ----
  init_dct_kernel<<<1, 256, 0, stream>>>(dctT);
  {
    long n = (long)kBS * kInputN * kInFeat;
    cvt_f16_kernel<<<(n + 255) / 256, 256, 0, stream>>>(src, src16, n, 1.0f);
  }
  {
    long n = (long)512 * 66 * 6;
    reorder_w6_kernel<<<(n + 255) / 256, 256, 0, stream>>>(Wk1, Bk1, 1e-3f);
    reorder_w6_kernel<<<(n + 255) / 256, 256, 0, stream>>>(Wq1, Bq1, 1e-3f);
  }
  {
    long n = (long)512 * 512 * 5;
    reorder_w5_kernel<<<(n + 255) / 256, 256, 0, stream>>>(Wk2, Bk2);
    reorder_w5_kernel<<<(n + 255) / 256, 256, 0, stream>>>(Wq2, Bq2);
  }
  cvt_f16_kernel<<<(40L * 512 + 255) / 256, 256, 0, stream>>>(gc1_W, Wg1, 40L * 512, 1.0f);
  cvt_f16_kernel<<<(4L * 512 * 512 + 255) / 256, 256, 0, stream>>>(gcb_W, Wgb, 4L * 512 * 512, 1.0f);
  cvt_f16_kernel<<<(512L * 40 + 255) / 256, 256, 0, stream>>>(gc7_W, Wg7, 512L * 40, 1.0f);

  // ---- convs as WMMA GEMMs (im2col = strided views, no copies) ----
  gemm_f16_kernel<true><<<dim3(80, 8), 256, 0, stream>>>(
      src16, 0L, 20, 66L, 50L * 66, 396, Bk1, 512, out1k, 512, 512);
  gemm_f16_kernel<true><<<dim3(20, 8), 256, 0, stream>>>(
      src16, 40L * 66, 5, 66L, 50L * 66, 396, Bq1, 512, out1q, 512, 512);
  gemm_f16_kernel<true><<<dim3(64, 8), 256, 0, stream>>>(
      out1k, 0L, 16, 512L, 20L * 512, 2560, Bk2, 512, keyf, 512, 512);
  gemm_f16_kernel<true><<<dim3(4, 8), 256, 0, stream>>>(
      out1q, 0L, 1, 512L, 5L * 512, 2560, Bq2, 512, qf, 512, 512);

  // ---- attention + x assembly ----
  score_att_kernel<<<kBS, 512, 0, stream>>>(qf, keyf, attb);
  build_x_kernel<<<kBS, 256, 0, stream>>>(src, attb, dctT, xf32, xf16);

  // ---- gc1 ----
  gemm_f16_kernel<false><<<dim3(264, 8), 256, 0, stream>>>(
      xf16, 0L, 66, (long)kGcnIn, 66L * kGcnIn, kGcnIn, Wg1, 512, Cbuf, 512, 512);
  gcn_att_kernel<<<dim3(4, kBS), 256, 0, stream>>>(
      Cbuf, gc1_at, gc1_b, bn1_g, bn1_b, nullptr, ybuf, bnscale);

  // ---- residual GCN stages ----
  const size_t ybytes = (size_t)kRowsGcn * kHid * 2;
  for (int st = 0; st < 2; ++st) {
    hipMemcpyAsync(y0buf, ybuf, ybytes, hipMemcpyDeviceToDevice, stream);
    for (int l = 0; l < 2; ++l) {
      const int li = st * 2 + l;
      gemm_f16_kernel<false><<<dim3(264, 8), 256, 0, stream>>>(
          ybuf, 0L, 66, (long)kHid, 66L * kHid, kHid, Wgb + (long)li * kHid * kHid, 512,
          Cbuf, 512, 512);
      gcn_att_kernel<<<dim3(4, kBS), 256, 0, stream>>>(
          Cbuf, gcb_at + (long)li * 66 * 66, gcb_b + (long)li * kHid,
          gcb_g + (long)li * 66 * kHid, gcb_be + (long)li * 66 * kHid,
          (l == 1) ? y0buf : nullptr, ybuf, bnscale);
    }
  }

  // ---- gc7 + x skip -> dct_out ----
  gemm_f16_kernel<false><<<dim3(264, 1), 256, 0, stream>>>(
      ybuf, 0L, 66, (long)kHid, 66L * kHid, kHid, Wg7, kGcnIn, C7, kGcnIn, kGcnIn);
  gcn_final_kernel<<<kBS, 256, 0, stream>>>(C7, gc7_at, gc7_b, xf32, dcto);

  // ---- inverse DCT -> output (512, 35, 1, 66) ----
  idct_kernel<<<kBS, 256, 0, stream>>>(dcto, dctT, out);
}
